// Gemma4VisionAttention_70626442216124
// MI455X (gfx1250) — matrix operand order repaired and checked by reading, not compiled, not run on hardware
//
#include <hip/hip_runtime.h>
#include <hip/hip_bf16.h>

// Problem constants (match reference)
#define BB   8
#define SS   1024
#define DD   1024
#define HH   16
#define KVH  16
#define HDIM 64
#define MROWS (BB*SS)   // 8192
#define NQKV 3072

typedef __attribute__((ext_vector_type(16))) __bf16 bf16x16;
typedef __attribute__((ext_vector_type(8)))  __bf16 bf16x8;
typedef __attribute__((ext_vector_type(8)))  float  floatx8;
typedef __attribute__((ext_vector_type(4)))  unsigned int uint4v;
typedef __attribute__((ext_vector_type(8)))  int int8v;
typedef __attribute__((ext_vector_type(4)))  int int4v;

#if defined(__has_builtin)
#if __has_builtin(__builtin_amdgcn_tensor_load_to_lds) && \
    __has_builtin(__builtin_amdgcn_s_wait_tensorcnt)
#define USE_TDM 1
#endif
#endif

static __device__ __forceinline__ unsigned short f2bf(float f) {
  union { __bf16 h; unsigned short u; } cv; cv.h = (__bf16)f; return cv.u;
}

// Raw LDS byte offset of a shared-memory object (addrspace(3) value).
typedef const __attribute__((address_space(3))) void* as3_cvp;
static __device__ __forceinline__ unsigned ldsoff(const void* p) {
  return (unsigned)(unsigned long long)(as3_cvp)p;
}

// Async global->LDS 16B copy (ASYNCcnt-tracked; bypasses VGPRs).
static __device__ __forceinline__ void async_load_b128(unsigned lds,
                                                       const void* gaddr) {
  asm volatile("global_load_async_to_lds_b128 %0, %1, off"
               :: "v"(lds), "v"(gaddr) : "memory");
}
static __device__ __forceinline__ void wait_asynccnt0() {
  asm volatile("s_wait_asynccnt 0x0" ::: "memory");
}

#ifdef USE_TDM
// TDM 2-D tile load: tileH rows x tileW elems (2B each), global row pitch
// strideElems, packed contiguously at LDS byte offset `lds`.
// 6-arg builtin form: (g0, g1, g2, g3, g4, cpol).
static __device__ __forceinline__ void tdm_load_2d(unsigned lds, const void* gptr,
                                                   unsigned tileW, unsigned tileH,
                                                   unsigned strideElems) {
  unsigned long long ga = (unsigned long long)gptr;
  const unsigned td0 = 0x10000u, td1 = 0x10000u;  // large tensor dims (no clip)
  uint4v g0;
  g0[0] = 1u;                                       // count=1 (user descriptor)
  g0[1] = lds;                                      // lds_addr (bytes)
  g0[2] = (unsigned)(ga & 0xffffffffu);             // global_addr[31:0]
  g0[3] = (unsigned)((ga >> 32) & 0x1ffffffu) | (2u << 30);  // [56:32] | type=2
  int8v g1;
  g1[0] = (int)(1u << 16);                          // data_size=1 (2 bytes)
  g1[1] = (int)((td0 & 0xffffu) << 16);             // tensor_dim0[15:0]
  g1[2] = (int)((td0 >> 16) | ((td1 & 0xffffu) << 16));
  g1[3] = (int)((td1 >> 16) | (tileW << 16));       // tile_dim0
  g1[4] = (int)(tileH & 0xffffu);                   // tile_dim1 (tile_dim2=0)
  g1[5] = (int)strideElems;                         // tensor_dim0_stride[31:0]
  g1[6] = 0;
  g1[7] = 0;
  int4v z4 = {0, 0, 0, 0};
  int8v z8 = {0, 0, 0, 0, 0, 0, 0, 0};
  __builtin_amdgcn_tensor_load_to_lds(g0, g1, z4, z4, z8, 0);
}
#endif

// ---------------- Kernel 1: f32 -> bf16 (hidden states) ----------------
__global__ void cvt_f32_bf16_kernel(const float* __restrict__ src,
                                    unsigned short* __restrict__ dst, int n4) {
  int idx = blockIdx.x * blockDim.x + threadIdx.x;
  if (idx >= n4) return;
  float4 v = ((const float4*)src)[idx];
  uint2 p;
  p.x = (unsigned)f2bf(v.x) | ((unsigned)f2bf(v.y) << 16);
  p.y = (unsigned)f2bf(v.z) | ((unsigned)f2bf(v.w) << 16);
  *(uint2*)(dst + 4 * (size_t)idx) = p;
}

// ------------- Kernel 2: transpose f32 [K][N] -> bf16 [N][K] -----------
__global__ void transpose_f32_bf16_kernel(const float* __restrict__ src,
                                          unsigned short* __restrict__ dst,
                                          int K, int N) {
  int idx = blockIdx.x * blockDim.x + threadIdx.x;
  if (idx >= K * N) return;
  int k = idx % K, n = idx / K;
  dst[(size_t)n * K + k] = f2bf(src[(size_t)k * N + n]);
}

// Build a 16-element bf16 fragment from two contiguous 16-byte chunks.
static __device__ __forceinline__ bf16x16 mkfrag(const unsigned short* lo,
                                                 const unsigned short* hi) {
  bf16x8 a = *(const bf16x8*)lo;
  bf16x8 b = *(const bf16x8*)hi;
  bf16x16 r;
#pragma unroll
  for (int i = 0; i < 8; ++i) { r[i] = a[i]; r[i + 8] = b[i]; }
  return r;
}

// --------- Kernel 3: fused QKV GEMM + RMS-norm + 2D-RoPE epilogue ------
// 64x64 C tile; K-step 64; double-buffered async LDS staging.
__global__ void qkv_gemm_kernel(const unsigned short* __restrict__ hsb,
                                const unsigned short* __restrict__ wT,
                                unsigned short* __restrict__ qb,
                                unsigned short* __restrict__ kbuf,
                                unsigned short* __restrict__ vT,
                                const float* __restrict__ cosp,
                                const float* __restrict__ sinp,
                                const float* __restrict__ qw,
                                const float* __restrict__ kw) {
  __shared__ __align__(16) unsigned short As[2][64][64];
  __shared__ __align__(16) unsigned short Bs[2][64][64];
  __shared__ float Cs[64][64];

  const int tid  = threadIdx.x;
  const int lane = tid & 31, wid = tid >> 5;
  const int wr = wid & 3, wc = wid >> 2;          // wave = 16 rows x 32 cols
  const int m0 = blockIdx.y * 64;
  const int nb = blockIdx.x;                      // 0..47 head-column blocks
  const int n0 = nb * 64;

  floatx8 acc0 = {}; floatx8 acc1 = {};

  const int lrow = tid >> 2, lcol = (tid & 3) * 16;   // cooperative staging
  const int kh   = (lane >> 4) * 16;
  const int arow = 16 * wr + (lane & 15);
  const int bc0  = 32 * wc + (lane & 15);
  const int bk   = (lane >> 4) * 16;

  const unsigned short* ag = hsb + (size_t)(m0 + lrow) * DD + lcol;
  const unsigned short* bg = wT + (size_t)(n0 + lrow) * DD + lcol;

  // Stage 0 prologue.
  async_load_b128(ldsoff(&As[0][lrow][lcol]),     ag);
  async_load_b128(ldsoff(&As[0][lrow][lcol + 8]), ag + 8);
  async_load_b128(ldsoff(&Bs[0][lrow][lcol]),     bg);
  async_load_b128(ldsoff(&Bs[0][lrow][lcol + 8]), bg + 8);

  for (int k0 = 0; k0 < DD; k0 += 64) {
    const int p = (k0 >> 6) & 1;
    wait_asynccnt0();              // my share of stage p is in LDS
    __syncthreads();               // everyone's share is in LDS
    if (k0 + 64 < DD) {            // prefetch stage p^1 while computing on p
      const unsigned short* ag2 = ag + k0 + 64;
      const unsigned short* bg2 = bg + k0 + 64;
      async_load_b128(ldsoff(&As[p ^ 1][lrow][lcol]),     ag2);
      async_load_b128(ldsoff(&As[p ^ 1][lrow][lcol + 8]), ag2 + 8);
      async_load_b128(ldsoff(&Bs[p ^ 1][lrow][lcol]),     bg2);
      async_load_b128(ldsoff(&Bs[p ^ 1][lrow][lcol + 8]), bg2 + 8);
      __builtin_prefetch(ag2 + 64, 0, 1);
      __builtin_prefetch(bg2 + 64, 0, 1);
    }
#pragma unroll
    for (int ks = 0; ks < 2; ++ks) {
      const int o = 32 * ks;
      bf16x16 af = mkfrag(&As[p][arow][o + kh], &As[p][arow][o + kh + 8]);
      bf16x16 b0 = mkfrag(&Bs[p][bc0][o + bk], &Bs[p][bc0][o + bk + 8]);
      bf16x16 b1 = mkfrag(&Bs[p][bc0 + 16][o + bk], &Bs[p][bc0 + 16][o + bk + 8]);
      acc0 = __builtin_amdgcn_wmma_f32_16x16x32_bf16(false, af, false, b0,
                                                     (short)0, acc0, false, false);
      acc1 = __builtin_amdgcn_wmma_f32_16x16x32_bf16(false, af, false, b1,
                                                     (short)0, acc1, false, false);
    }
  }

  // Spill C accumulators to LDS (C layout: VGPR r -> rows r / r+8 per lane half)
#pragma unroll
  for (int r = 0; r < 8; ++r) {
    int row = 16 * wr + r + 8 * (lane >> 4);
    Cs[row][32 * wc + (lane & 15)]      = acc0[r];
    Cs[row][32 * wc + 16 + (lane & 15)] = acc1[r];
  }
  __syncthreads();

  // Epilogue: 4 threads per row, 16 head-dims each.
  const int row = tid >> 2;
  const int u   = tid & 3;
  const int m   = m0 + row;
  const int bidx = m / SS, sidx = m % SS;

  float ss = 0.f;
#pragma unroll
  for (int j = 0; j < 16; ++j) { float v = Cs[row][16 * u + j]; ss += v * v; }
  ss += __shfl_xor(ss, 1, 32);
  ss += __shfl_xor(ss, 2, 32);
  const float rnorm = rsqrtf(ss * (1.0f / 64.0f) + 1e-6f);

  if (nb < 32) {                                   // Q or K: weight + RoPE
    const float* w = (nb < 16) ? qw : kw;
    unsigned short* dst = (nb < 16) ? qb : kbuf;
    const int h = (nb < 16) ? nb : nb - 16;
    const float* cp = cosp + (size_t)m * HDIM;
    const float* sp = sinp + (size_t)m * HDIM;
#pragma unroll
    for (int j = 0; j < 16; ++j) {
      int hd = 16 * u + j;
      int pc = hd ^ 16;                            // rotate-half pair in chunk
      float x  = Cs[row][hd] * rnorm * w[hd];
      float xp = Cs[row][pc] * rnorm * w[pc];
      float rot = (hd & 16) ? xp : -xp;
      float o = x * cp[hd] + rot * sp[hd];
      dst[(size_t)m * (HH * HDIM) + h * HDIM + hd] = f2bf(o);
    }
  } else {                                         // V: norm only, store V^T
    const int h = nb - 32;
#pragma unroll
    for (int j = 0; j < 16; ++j) {
      int hd = 16 * u + j;
      vT[(((size_t)bidx * KVH + h) * HDIM + hd) * SS + sidx] =
          f2bf(Cs[row][hd] * rnorm);
    }
  }
}

// -------- Kernel 4: flash attention (online softmax, WMMA bf16) --------
// K/V tiles delivered by the Tensor Data Mover when available.
__global__ void attn_kernel(const unsigned short* __restrict__ qb,
                            const unsigned short* __restrict__ kbuf,
                            const unsigned short* __restrict__ vT,
                            unsigned short* __restrict__ attnb) {
  __shared__ __align__(16) unsigned short Ksh[32][64];   // [key][hd]
  __shared__ __align__(16) unsigned short Vsh[64][32];   // [hd][key]
  __shared__ __align__(16) unsigned short Psh[8][16][32];

  const int tid = threadIdx.x, lane = tid & 31, wid = tid >> 5;
  const int b = blockIdx.y >> 4, h = blockIdx.y & 15;
  const int s0 = blockIdx.x * 128 + wid * 16;            // wave's 16 q-rows
  const int kh   = (lane >> 4) * 16;
  const int arow = lane & 15;
  const int bk   = (lane >> 4) * 16;

  // Q fragments (HD=64 -> two K=32 fragments), loaded once.
  const unsigned short* qp =
      qb + (size_t)(b * SS + s0 + arow) * (HH * HDIM) + h * HDIM;
  bf16x16 qf0 = mkfrag(qp + kh,      qp + kh + 8);
  bf16x16 qf1 = mkfrag(qp + 32 + kh, qp + 40 + kh);

  floatx8 O0 = {}, O1 = {}, O2 = {}, O3 = {};
  float mrow[8], lsum[8];
#pragma unroll
  for (int r = 0; r < 8; ++r) { mrow[r] = -1e30f; lsum[r] = 0.f; }

#ifndef USE_TDM
  const int klrow = tid >> 3, klchunk = (tid & 7) * 8;   // 32x64 K tile
  const int vlrow = tid >> 2, vlchunk = (tid & 3) * 8;   // 64x32 V^T tile
#endif

  for (int c = 0; c < SS; c += 32) {
    __syncthreads();                 // previous tile fully consumed
#ifdef USE_TDM
    if (wid == 0) {
      tdm_load_2d(ldsoff(&Ksh[0][0]),
                  kbuf + (size_t)(b * SS + c) * (KVH * HDIM) + h * HDIM,
                  64, 32, KVH * HDIM);
      tdm_load_2d(ldsoff(&Vsh[0][0]),
                  vT + ((size_t)b * KVH + h) * HDIM * SS + c,
                  32, 64, SS);
      __builtin_amdgcn_s_wait_tensorcnt(0);
    }
#else
    *(uint4*)&Ksh[klrow][klchunk] = *(const uint4*)(
        kbuf + (size_t)(b * SS + c + klrow) * (KVH * HDIM) + h * HDIM + klchunk);
    *(uint4*)&Vsh[vlrow][vlchunk] = *(const uint4*)(
        vT + (((size_t)b * KVH + h) * HDIM + vlrow) * SS + c + vlchunk);
#endif
    __syncthreads();

    // scores: 16 q-rows x 32 keys
    floatx8 sa0 = {}, sa1 = {};
    {
      bf16x16 k0 = mkfrag(&Ksh[arow][bk],      &Ksh[arow][bk + 8]);
      bf16x16 k1 = mkfrag(&Ksh[16 + arow][bk], &Ksh[16 + arow][bk + 8]);
      sa0 = __builtin_amdgcn_wmma_f32_16x16x32_bf16(false, qf0, false, k0,
                                                    (short)0, sa0, false, false);
      sa1 = __builtin_amdgcn_wmma_f32_16x16x32_bf16(false, qf0, false, k1,
                                                    (short)0, sa1, false, false);
      bf16x16 k2 = mkfrag(&Ksh[arow][32 + bk],      &Ksh[arow][32 + bk + 8]);
      bf16x16 k3 = mkfrag(&Ksh[16 + arow][32 + bk], &Ksh[16 + arow][32 + bk + 8]);
      sa0 = __builtin_amdgcn_wmma_f32_16x16x32_bf16(false, qf1, false, k2,
                                                    (short)0, sa0, false, false);
      sa1 = __builtin_amdgcn_wmma_f32_16x16x32_bf16(false, qf1, false, k3,
                                                    (short)0, sa1, false, false);
    }

    // online softmax per row; xor masks < 16 stay within the 16-lane half
#pragma unroll
    for (int r = 0; r < 8; ++r) {
      float vmax = fmaxf(sa0[r], sa1[r]);
      vmax = fmaxf(vmax, __shfl_xor(vmax, 1, 32));
      vmax = fmaxf(vmax, __shfl_xor(vmax, 2, 32));
      vmax = fmaxf(vmax, __shfl_xor(vmax, 4, 32));
      vmax = fmaxf(vmax, __shfl_xor(vmax, 8, 32));
      float mnew  = fmaxf(mrow[r], vmax);
      float scale = __expf(mrow[r] - mnew);
      float p0 = __expf(sa0[r] - mnew);
      float p1 = __expf(sa1[r] - mnew);
      float rs = p0 + p1;
      rs += __shfl_xor(rs, 1, 32);
      rs += __shfl_xor(rs, 2, 32);
      rs += __shfl_xor(rs, 4, 32);
      rs += __shfl_xor(rs, 8, 32);
      lsum[r] = lsum[r] * scale + rs;
      mrow[r] = mnew;
      O0[r] *= scale; O1[r] *= scale; O2[r] *= scale; O3[r] *= scale;
      int rr = r + 8 * (lane >> 4);
      Psh[wid][rr][lane & 15]        = f2bf(p0);
      Psh[wid][rr][16 + (lane & 15)] = f2bf(p1);
    }

    // P @ V : A = P (16x32 keys), B = V (32 keys x 16 hd per acc)
    bf16x16 pf = mkfrag(&Psh[wid][arow][kh], &Psh[wid][arow][kh + 8]);
    bf16x16 v0 = mkfrag(&Vsh[arow][bk],      &Vsh[arow][bk + 8]);
    bf16x16 v1 = mkfrag(&Vsh[16 + arow][bk], &Vsh[16 + arow][bk + 8]);
    bf16x16 v2 = mkfrag(&Vsh[32 + arow][bk], &Vsh[32 + arow][bk + 8]);
    bf16x16 v3 = mkfrag(&Vsh[48 + arow][bk], &Vsh[48 + arow][bk + 8]);
    O0 = __builtin_amdgcn_wmma_f32_16x16x32_bf16(false, pf, false, v0,
                                                 (short)0, O0, false, false);
    O1 = __builtin_amdgcn_wmma_f32_16x16x32_bf16(false, pf, false, v1,
                                                 (short)0, O1, false, false);
    O2 = __builtin_amdgcn_wmma_f32_16x16x32_bf16(false, pf, false, v2,
                                                 (short)0, O2, false, false);
    O3 = __builtin_amdgcn_wmma_f32_16x16x32_bf16(false, pf, false, v3,
                                                 (short)0, O3, false, false);
  }

  // normalize and write attn output (bf16) for the final projection
#pragma unroll
  for (int r = 0; r < 8; ++r) {
    float inv = 1.0f / lsum[r];
    int rr = r + 8 * (lane >> 4);
    size_t mg = (size_t)(b * SS + s0 + rr);
    unsigned short* dp = attnb + mg * (HH * HDIM) + h * HDIM;
    dp[(lane & 15)]      = f2bf(O0[r] * inv);
    dp[16 + (lane & 15)] = f2bf(O1[r] * inv);
    dp[32 + (lane & 15)] = f2bf(O2[r] * inv);
    dp[48 + (lane & 15)] = f2bf(O3[r] * inv);
  }
}

// ----------------- Kernel 5: output projection GEMM --------------------
__global__ void out_gemm_kernel(const unsigned short* __restrict__ ab,
                                const unsigned short* __restrict__ wT,
                                float* __restrict__ out) {
  __shared__ __align__(16) unsigned short As[2][64][64];
  __shared__ __align__(16) unsigned short Bs[2][64][64];

  const int tid = threadIdx.x, lane = tid & 31, wid = tid >> 5;
  const int wr = wid & 3, wc = wid >> 2;
  const int m0 = blockIdx.y * 64;
  const int n0 = blockIdx.x * 64;

  floatx8 acc0 = {}; floatx8 acc1 = {};
  const int lrow = tid >> 2, lcol = (tid & 3) * 16;
  const int kh   = (lane >> 4) * 16;
  const int arow = 16 * wr + (lane & 15);
  const int bc0  = 32 * wc + (lane & 15);
  const int bk   = (lane >> 4) * 16;

  const unsigned short* ag = ab + (size_t)(m0 + lrow) * DD + lcol;
  const unsigned short* bg = wT + (size_t)(n0 + lrow) * DD + lcol;

  async_load_b128(ldsoff(&As[0][lrow][lcol]),     ag);
  async_load_b128(ldsoff(&As[0][lrow][lcol + 8]), ag + 8);
  async_load_b128(ldsoff(&Bs[0][lrow][lcol]),     bg);
  async_load_b128(ldsoff(&Bs[0][lrow][lcol + 8]), bg + 8);

  for (int k0 = 0; k0 < DD; k0 += 64) {
    const int p = (k0 >> 6) & 1;
    wait_asynccnt0();
    __syncthreads();
    if (k0 + 64 < DD) {
      const unsigned short* ag2 = ag + k0 + 64;
      const unsigned short* bg2 = bg + k0 + 64;
      async_load_b128(ldsoff(&As[p ^ 1][lrow][lcol]),     ag2);
      async_load_b128(ldsoff(&As[p ^ 1][lrow][lcol + 8]), ag2 + 8);
      async_load_b128(ldsoff(&Bs[p ^ 1][lrow][lcol]),     bg2);
      async_load_b128(ldsoff(&Bs[p ^ 1][lrow][lcol + 8]), bg2 + 8);
      __builtin_prefetch(ag2 + 64, 0, 1);
      __builtin_prefetch(bg2 + 64, 0, 1);
    }
#pragma unroll
    for (int ks = 0; ks < 2; ++ks) {
      const int o = 32 * ks;
      bf16x16 af = mkfrag(&As[p][arow][o + kh], &As[p][arow][o + kh + 8]);
      bf16x16 b0 = mkfrag(&Bs[p][bc0][o + bk], &Bs[p][bc0][o + bk + 8]);
      bf16x16 b1 = mkfrag(&Bs[p][bc0 + 16][o + bk], &Bs[p][bc0 + 16][o + bk + 8]);
      acc0 = __builtin_amdgcn_wmma_f32_16x16x32_bf16(false, af, false, b0,
                                                     (short)0, acc0, false, false);
      acc1 = __builtin_amdgcn_wmma_f32_16x16x32_bf16(false, af, false, b1,
                                                     (short)0, acc1, false, false);
    }
  }

#pragma unroll
  for (int r = 0; r < 8; ++r) {
    int row = m0 + 16 * wr + r + 8 * (lane >> 4);
    int col = n0 + 32 * wc + (lane & 15);
    out[(size_t)row * DD + col]      = acc0[r];
    out[(size_t)row * DD + col + 16] = acc1[r];
  }
}

// ------------------------------- launch --------------------------------
extern "C" void kernel_launch(void* const* d_in, const int* in_sizes, int n_in,
                              void* d_out, int out_size, void* d_ws, size_t ws_size,
                              hipStream_t stream) {
  (void)in_sizes; (void)n_in; (void)out_size; (void)ws_size;
  const float* hs   = (const float*)d_in[0];
  const float* cosp = (const float*)d_in[1];
  const float* sinp = (const float*)d_in[2];
  // d_in[3] = position_ids: reference uses only its last-dim size (NDIM=2)
  const float* wq = (const float*)d_in[4];
  const float* wk = (const float*)d_in[5];
  const float* wv = (const float*)d_in[6];
  const float* wo = (const float*)d_in[7];
  const float* qw = (const float*)d_in[8];
  const float* kw = (const float*)d_in[9];
  float* out = (float*)d_out;

  char* ws = (char*)d_ws;
  unsigned short* hsb   = (unsigned short*)(ws + ((size_t)0  << 20)); // 16 MB
  unsigned short* wqkvT = (unsigned short*)(ws + ((size_t)16 << 20)); //  6 MB
  unsigned short* woT   = (unsigned short*)(ws + ((size_t)22 << 20)); //  2 MB
  unsigned short* qb    = (unsigned short*)(ws + ((size_t)24 << 20)); // 16 MB
  unsigned short* kbuf  = (unsigned short*)(ws + ((size_t)40 << 20)); // 16 MB
  unsigned short* vT    = (unsigned short*)(ws + ((size_t)56 << 20)); // 16 MB
  unsigned short* attnb = (unsigned short*)(ws + ((size_t)72 << 20)); // 16 MB

  const int n4 = MROWS * DD / 4;
  cvt_f32_bf16_kernel<<<(n4 + 255) / 256, 256, 0, stream>>>(hs, hsb, n4);

  const int tblk = (1024 * 1024) / 256;
  transpose_f32_bf16_kernel<<<tblk, 256, 0, stream>>>(wq, wqkvT, 1024, 1024);
  transpose_f32_bf16_kernel<<<tblk, 256, 0, stream>>>(wk, wqkvT + 1024 * 1024, 1024, 1024);
  transpose_f32_bf16_kernel<<<tblk, 256, 0, stream>>>(wv, wqkvT + 2 * 1024 * 1024, 1024, 1024);
  transpose_f32_bf16_kernel<<<tblk, 256, 0, stream>>>(wo, woT, 1024, 1024);

  qkv_gemm_kernel<<<dim3(48, 128), 256, 0, stream>>>(hsb, wqkvT, qb, kbuf, vT,
                                                     cosp, sinp, qw, kw);
  attn_kernel<<<dim3(8, 128), 256, 0, stream>>>(qb, kbuf, vT, attnb);
  out_gemm_kernel<<<dim3(16, 128), 256, 0, stream>>>(attnb, woT, out);
}
